// BevEncode_45414984188204
// MI455X (gfx1250) — compile-verified, matrix operand order checked by reading, not run on hardware
//
#include <hip/hip_runtime.h>
#include <math.h>

// ---------------- Problem constants ----------------
#define BB   2
#define CC   128
#define HH   256
#define WW   256
#define HO   128
#define WO   128
#define GG   64
#define KK   9          // 3x3 taps
#define NOFF 1152       // G*2*K offset channels
#define NMSK 576        // G*K mask channels
#define NTOT 1728       // NOFF+NMSK
#define NPAD 1792       // padded to 14 * 128 (rows >= NTOT are zero-filled)
#define KDIM 1152       // C * 9, ordered k = tap*128 + c
#define MTOT 32768      // B*HO*WO

typedef __attribute__((ext_vector_type(16))) __bf16 v16bf;
typedef __attribute__((ext_vector_type(8)))  float  v8f;
typedef __attribute__((ext_vector_type(4)))  int    v4i_t;

// ---- CDNA5 async global->LDS support probe ----
#if defined(__has_builtin)
#  if __has_builtin(__builtin_amdgcn_global_load_async_to_lds_b128)
#    define HAVE_ASYNC 1
#  endif
#endif
#ifndef HAVE_ASYNC
#  define HAVE_ASYNC 0
#endif

#if HAVE_ASYNC
template <int N>
__device__ __forceinline__ void wait_asynccnt() {
#  if defined(__has_builtin) && __has_builtin(__builtin_amdgcn_s_wait_asynccnt)
    __builtin_amdgcn_s_wait_asynccnt(N);
#  else
    asm volatile("s_wait_asynccnt %0" :: "n"(N));
#  endif
}

__device__ __forceinline__ void async_b128(const unsigned short* g, unsigned short* l) {
    __builtin_amdgcn_global_load_async_to_lds_b128((v4i_t*)g, (v4i_t*)l, 0, 0);
}
#endif

__device__ __forceinline__ unsigned short f2bf(float f) {
    unsigned int u = __float_as_uint(f);
    unsigned int r = u + 0x7FFFu + ((u >> 16) & 1u);   // round-to-nearest-even
    return (unsigned short)(r >> 16);
}

// ---------------- 1) x (NCHW f32) -> xh (NHWC bf16), tiled transpose ----------------
__global__ __launch_bounds__(256) void pack_x_kernel(const float* __restrict__ x,
                                                     unsigned short* __restrict__ xh) {
    __shared__ unsigned short t[32 * 130];   // 32 w-cols x 128 channels, +2 pad
    const int blk = blockIdx.x;              // B * H * (W/32) = 4096
    const int w0  = (blk & 7) << 5;
    const int h   = (blk >> 3) & 255;
    const int b   = blk >> 11;
    const int tid = threadIdx.x;
#pragma unroll
    for (int i = 0; i < 16; ++i) {
        int idx = tid + i * 256;             // 0..4095
        int c  = idx >> 5;
        int wl = idx & 31;
        float v = x[(((size_t)b * CC + c) * HH + h) * WW + (w0 + wl)];
        t[wl * 130 + c] = f2bf(v);
    }
    __syncthreads();
#pragma unroll
    for (int i = 0; i < 16; ++i) {
        int idx = tid + i * 256;
        int wl = idx >> 7;
        int c  = idx & 127;
        xh[(((size_t)b * HH + h) * WW + (w0 + wl)) * CC + c] = t[wl * 130 + c];
    }
}

// ---------------- 2) pack weights: wp[n][k], k = tap*128 + c, n padded to 1792 ----------------
__global__ __launch_bounds__(256) void pack_w_kernel(const float* __restrict__ w_off,
                                                     const float* __restrict__ w_msk,
                                                     unsigned short* __restrict__ wp) {
    long i = (long)blockIdx.x * 256 + threadIdx.x;
    if (i >= (long)NPAD * KDIM) return;
    int k = (int)(i % KDIM);
    int n = (int)(i / KDIM);
    int tap = k >> 7;
    int c   = k & 127;
    float v = 0.f;
    if (n < NOFF)       v = w_off[((size_t)n * CC + c) * KK + tap];
    else if (n < NTOT)  v = w_msk[((size_t)(n - NOFF) * CC + c) * KK + tap];
    wp[i] = f2bf(v);
}

// ---------------- 3) implicit-GEMM conv via WMMA bf16 ----------------
// Tile: 128(M pixels) x 128(N channels) per workgroup; 8 waves; wave = 32x64 (2x4 wmma).
// Double-buffered LDS + async global->LDS copies (ASYNCcnt) when available.
#define ASTR 40   // LDS row stride in bf16 elements (80B: 16B aligned, bank-staggered)

__global__ __launch_bounds__(256) void gemm_conv_kernel(const unsigned short* __restrict__ xh,
                                                        const unsigned short* __restrict__ wp,
                                                        const float* __restrict__ b_off,
                                                        const float* __restrict__ b_msk,
                                                        float* __restrict__ offs,
                                                        float* __restrict__ maskb) {
    __shared__ alignas(16) unsigned short As[2][128 * ASTR];
    __shared__ alignas(16) unsigned short Bs[2][128 * ASTR];

    const int tid   = threadIdx.x;
    const int ntile = blockIdx.x;      // 0..13
    const int mtile = blockIdx.y;      // 0..255
    const int lane  = tid & 31;
    const int wv    = tid >> 5;        // 0..7
    const int mw    = wv >> 1;         // 0..3  (32 rows each)
    const int nw    = wv & 1;          // 0..1  (64 cols each)

    // cooperative load assignment: 2 threads per tile row, 32B each (2 x b128)
    const int arow  = tid >> 1;        // 0..127
    const int ahalf = tid & 1;         // 0 or 1 (16 bf16 each)

    // im2col source pixel for A row `arow`
    const int gm  = mtile * 128 + arow;
    const int b   = gm >> 14;
    const int pix = gm & 16383;
    const int ho  = pix >> 7;
    const int wo  = pix & 127;
    const int gn_row = ntile * 128 + arow;   // B row (always < NPAD)

    // 32B of guaranteed zeros (zero-padded tail row of wp) for OOB im2col taps
    const unsigned short* zsrc = wp + (size_t)(NPAD - 1) * KDIM + (ahalf << 4);

    v8f acc[2][4];
#pragma unroll
    for (int r = 0; r < 2; ++r)
#pragma unroll
        for (int c = 0; c < 4; ++c) acc[r][c] = (v8f){0.f,0.f,0.f,0.f,0.f,0.f,0.f,0.f};

    union Frag { uint4 q[2]; v16bf v; };

    const int khalf = (lane >> 4) << 4;  // element offset 0 / 16 along K
    const int lrow  = lane & 15;

    // source address of this thread's 32B slice of the A tile for chunk kcn
    auto a_src = [&](int kcn) -> const unsigned short* {
        const int tap = kcn >> 2;
        const int c0  = (kcn & 3) << 5;
        const int kh  = tap / 3;
        const int kw  = tap - kh * 3;
        const int iy  = ho * 2 - 1 + kh;
        const int ix  = wo * 2 - 1 + kw;
        if ((unsigned)iy < (unsigned)HH && (unsigned)ix < (unsigned)WW)
            return xh + ((((size_t)b * HH + iy) * WW + ix) << 7) + c0 + (ahalf << 4);
        return zsrc;
    };

#if HAVE_ASYNC
    // ---- async double-buffered pipeline: exactly 4 async b128 per thread per chunk ----
    auto issue = [&](int kcn, int sel) {
        const unsigned short* ga = a_src(kcn);
        const unsigned short* gb = wp + (size_t)gn_row * KDIM + (kcn << 5) + (ahalf << 4);
        unsigned short* la = &As[sel][arow * ASTR + (ahalf << 4)];
        unsigned short* lb = &Bs[sel][arow * ASTR + (ahalf << 4)];
        async_b128(ga,     la);
        async_b128(ga + 8, la + 8);
        async_b128(gb,     lb);
        async_b128(gb + 8, lb + 8);
    };

    issue(0, 0);
    for (int kc = 0; kc < 36; ++kc) {
        const int sel = kc & 1;
        if (kc + 1 < 36) {
            issue(kc + 1, sel ^ 1);
            wait_asynccnt<4>();      // chunk kc's 4 loads retired (in-order completion)
        } else {
            wait_asynccnt<0>();
        }
        __syncthreads();

        Frag af[2], bf[4];
#pragma unroll
        for (int r = 0; r < 2; ++r) {
            const uint4* p = (const uint4*)&As[sel][(mw * 32 + r * 16 + lrow) * ASTR + khalf];
            af[r].q[0] = p[0]; af[r].q[1] = p[1];
        }
#pragma unroll
        for (int c = 0; c < 4; ++c) {
            const uint4* p = (const uint4*)&Bs[sel][(nw * 64 + c * 16 + lrow) * ASTR + khalf];
            bf[c].q[0] = p[0]; bf[c].q[1] = p[1];
        }
#pragma unroll
        for (int r = 0; r < 2; ++r)
#pragma unroll
            for (int c = 0; c < 4; ++c)
                acc[r][c] = __builtin_amdgcn_wmma_f32_16x16x32_bf16(
                    false, af[r].v, false, bf[c].v, (short)0, acc[r][c], false, false);

        __syncthreads();             // reads done before this buffer is refilled
    }
#else
    // ---- fallback: synchronous staging (round-1 path), single buffer ----
    for (int kc = 0; kc < 36; ++kc) {
        const unsigned short* ga = a_src(kc);
        const uint4* gp = (const uint4*)ga;
        uint4 a0 = gp[0], a1 = gp[1];
        const uint4* bp = (const uint4*)(wp + (size_t)gn_row * KDIM + (kc << 5) + (ahalf << 4));
        uint4 b0 = bp[0], b1 = bp[1];

        __syncthreads();
        {
            uint4* d = (uint4*)&As[0][arow * ASTR + (ahalf << 4)];
            d[0] = a0; d[1] = a1;
            uint4* e = (uint4*)&Bs[0][arow * ASTR + (ahalf << 4)];
            e[0] = b0; e[1] = b1;
        }
        __syncthreads();

        Frag af[2], bf[4];
#pragma unroll
        for (int r = 0; r < 2; ++r) {
            const uint4* p = (const uint4*)&As[0][(mw * 32 + r * 16 + lrow) * ASTR + khalf];
            af[r].q[0] = p[0]; af[r].q[1] = p[1];
        }
#pragma unroll
        for (int c = 0; c < 4; ++c) {
            const uint4* p = (const uint4*)&Bs[0][(nw * 64 + c * 16 + lrow) * ASTR + khalf];
            bf[c].q[0] = p[0]; bf[c].q[1] = p[1];
        }
#pragma unroll
        for (int r = 0; r < 2; ++r)
#pragma unroll
            for (int c = 0; c < 4; ++c)
                acc[r][c] = __builtin_amdgcn_wmma_f32_16x16x32_bf16(
                    false, af[r].v, false, bf[c].v, (short)0, acc[r][c], false, false);
    }
#endif

    // epilogue: +bias, sigmoid on mask half, NCHW scatter
    const int lhi = lane >> 4;
#pragma unroll
    for (int c = 0; c < 4; ++c) {
        const int gn = ntile * 128 + nw * 64 + c * 16 + lrow;
        if (gn >= NTOT) continue;
        const bool is_mask = (gn >= NOFF);
        const float bias = is_mask ? b_msk[gn - NOFF] : b_off[gn];
#pragma unroll
        for (int r = 0; r < 2; ++r) {
#pragma unroll
            for (int j = 0; j < 8; ++j) {
                const int ml  = mw * 32 + r * 16 + j + (lhi << 3);
                const int gm2 = mtile * 128 + ml;
                const int bb  = gm2 >> 14;
                const int pp  = gm2 & 16383;
                float v = acc[r][c][j] + bias;
                if (is_mask) {
                    maskb[((size_t)bb * NMSK + (gn - NOFF)) * 16384 + pp] =
                        1.f / (1.f + __expf(-v));
                } else {
                    offs[((size_t)bb * NOFF + gn) * 16384 + pp] = v;
                }
            }
        }
    }
}

// ---------------- 4) deformable gather + grouped 3x3 conv (Cg=2 -> 2 outputs) ----------------
__global__ __launch_bounds__(256) void deform_kernel(const float* __restrict__ x,
                                                     const float* __restrict__ offs,
                                                     const float* __restrict__ maskb,
                                                     const float* __restrict__ wd,
                                                     float* __restrict__ out) {
    const int idx = blockIdx.x * 256 + threadIdx.x;   // < 2*64*128*128
    const int wo = idx & 127;
    const int ho = (idx >> 7) & 127;
    const int g  = (idx >> 14) & 63;
    const int b  = idx >> 20;

    const float* xc0 = x + (((size_t)b * CC + g * 2) * HH) * WW;
    const float* xc1 = xc0 + (size_t)HH * WW;
    const size_t spix  = ((size_t)ho << 7) + wo;
    const size_t obase = ((size_t)b * NOFF + (size_t)g * 18) * 16384 + spix;
    const size_t mbase = ((size_t)b * NMSK + (size_t)g * 9)  * 16384 + spix;

    float acc0 = 0.f, acc1 = 0.f;
#pragma unroll
    for (int k = 0; k < 9; ++k) {
        const float dy = offs[obase + (size_t)(2 * k)     * 16384];
        const float dx = offs[obase + (size_t)(2 * k + 1) * 16384];
        const float mk = maskb[mbase + (size_t)k * 16384];
        const float py = dy + (float)(ho * 2 - 1 + (k / 3));
        const float px = dx + (float)(wo * 2 - 1 + (k % 3));
        const float y0 = floorf(py), x0 = floorf(px);
        const float wy = py - y0, wx = px - x0;
        const int iy0 = (int)y0, ix0 = (int)x0;

        float v0 = 0.f, v1 = 0.f;
#pragma unroll
        for (int dyi = 0; dyi < 2; ++dyi) {
#pragma unroll
            for (int dxi = 0; dxi < 2; ++dxi) {
                const int yy = iy0 + dyi;
                const int xx = ix0 + dxi;
                if ((unsigned)yy < (unsigned)HH && (unsigned)xx < (unsigned)WW) {
                    const float wgt = (dyi ? wy : 1.f - wy) * (dxi ? wx : 1.f - wx);
                    const int id = yy * WW + xx;
                    v0 += wgt * xc0[id];
                    v1 += wgt * xc1[id];
                }
            }
        }
        v0 *= mk; v1 *= mk;
        acc0 += v0 * wd[((size_t)g * 4 + 0) * 9 + k] + v1 * wd[((size_t)g * 4 + 1) * 9 + k];
        acc1 += v0 * wd[((size_t)g * 4 + 2) * 9 + k] + v1 * wd[((size_t)g * 4 + 3) * 9 + k];
    }
    out[(((size_t)b * CC + g * 2    ) * 16384) + spix] = acc0;
    out[(((size_t)b * CC + g * 2 + 1) * 16384) + spix] = acc1;
}

// ---------------- launcher ----------------
extern "C" void kernel_launch(void* const* d_in, const int* in_sizes, int n_in,
                              void* d_out, int out_size, void* d_ws, size_t ws_size,
                              hipStream_t stream) {
    const float* x        = (const float*)d_in[0];
    const float* w_offset = (const float*)d_in[1];
    const float* b_offset = (const float*)d_in[2];
    const float* w_mask   = (const float*)d_in[3];
    const float* b_mask   = (const float*)d_in[4];
    const float* w_deform = (const float*)d_in[5];

    char* ws = (char*)d_ws;
    const size_t SZ_XH   = (size_t)BB * HH * WW * CC * 2;          // 32 MB
    const size_t SZ_WP   = (size_t)NPAD * KDIM * 2;                // ~4 MB
    const size_t SZ_OFFS = (size_t)BB * NOFF * HO * WO * 4;        // 144 MB
    unsigned short* xh    = (unsigned short*)ws;
    unsigned short* wpack = (unsigned short*)(ws + SZ_XH);
    float*          offs  = (float*)(ws + SZ_XH + SZ_WP);
    float*          maskb = (float*)(ws + SZ_XH + SZ_WP + SZ_OFFS);

    pack_x_kernel<<<BB * HH * (WW / 32), 256, 0, stream>>>(x, xh);
    pack_w_kernel<<<(int)(((long)NPAD * KDIM + 255) / 256), 256, 0, stream>>>(w_offset, w_mask, wpack);

    dim3 grid(NPAD / 128, MTOT / 128);   // 14 x 256
    gemm_conv_kernel<<<grid, 256, 0, stream>>>(xh, wpack, b_offset, b_mask, offs, maskb);

    deform_kernel<<<(BB * GG * HO * WO) / 256, 256, 0, stream>>>(x, offs, maskb, w_deform, (float*)d_out);
}
